// GRUModel_87162066305732
// MI455X (gfx1250) — matrix operand order repaired and checked by reading, not compiled, not run on hardware
//
#include <hip/hip_runtime.h>
#include <hip/hip_bf16.h>

typedef __attribute__((ext_vector_type(16))) _Float16 v16h;
typedef __attribute__((ext_vector_type(8)))  _Float16 v8h;
typedef __attribute__((ext_vector_type(8)))  float    v8f;

#define T_SEQ 2048
#define LDH   72      // padded LDS row stride (halves); rows stay 16B-aligned

#if __has_builtin(__builtin_amdgcn_tanh_f32)
__device__ __forceinline__ float dtanh(float x) {
  return __builtin_amdgcn_tanh_f32(x);     // gfx1250 v_tanh_f32
}
#else
__device__ __forceinline__ float dtanh(float x) {
  float e = __expf(2.0f * x);
  return __fdividef(e - 1.0f, e + 1.0f);
}
#endif
__device__ __forceinline__ float sigf(float x) {
  return fmaf(0.5f, dtanh(0.5f * x), 0.5f);   // sigmoid via tanh
}
__device__ __forceinline__ v8f wmma16(v16h a, v16h b, v8f c) {
  // D = A(16x32 f16) * B(32x16 f16) + C(16x16 f32)
  return __builtin_amdgcn_wmma_f32_16x16x32_f16(false, a, false, b, (short)0, c,
                                                false, false);
}

__global__ __launch_bounds__(128) void gru2_fused_wmma(
    const float* __restrict__ x,
    const float* __restrict__ Wih0, const float* __restrict__ Whh0,
    const float* __restrict__ bih0, const float* __restrict__ bhh0,
    const float* __restrict__ Wih1, const float* __restrict__ Whh1,
    const float* __restrict__ bih1, const float* __restrict__ bhh1,
    const float* __restrict__ fcw,  const float* __restrict__ fcb,
    float* __restrict__ out)
{
  // ping-pong state buffers: write step t into [t&1], read t-1 from [t^1]
  __shared__ __align__(16) _Float16 lds_h1[2][16 * LDH];
  __shared__ __align__(16) _Float16 lds_h2[2][16 * LDH];
  __shared__ float xbuf[2][32];
  __shared__ float outacc[2][16];

  const int tid  = threadIdx.x;
  const int wav  = tid >> 5;
  const int lane = tid & 31;
  const int hf   = lane >> 4;     // lane half (0/1)
  const int lcol = lane & 15;
  const int j    = wav * 16 + lcol;   // hidden/gate column owned by this lane
  const int m0   = blockIdx.x * 16;   // batch-row base for this workgroup

  // ---- zero LDS state (h0 = 0) ----
  for (int i = tid; i < 16 * LDH; i += 128) {
    lds_h1[0][i] = (_Float16)0.f;  lds_h1[1][i] = (_Float16)0.f;
    lds_h2[0][i] = (_Float16)0.f;  lds_h2[1][i] = (_Float16)0.f;
  }
  if (tid < 32) outacc[tid >> 4][tid & 15] = 0.f;

  // ---- per-lane scalar constants (layer0 input proj + fused biases) ----
  const float wr0 = Wih0[j * 2 + 0],         wr1 = Wih0[j * 2 + 1];
  const float wz0 = Wih0[(64 + j) * 2 + 0],  wz1 = Wih0[(64 + j) * 2 + 1];
  const float wn0 = Wih0[(128 + j) * 2 + 0], wn1 = Wih0[(128 + j) * 2 + 1];
  const float c0r  = bih0[j]        + bhh0[j];
  const float c0z  = bih0[64 + j]   + bhh0[64 + j];
  const float c0xn = bih0[128 + j];
  const float c0hn = bhh0[128 + j];
  const float c1r  = bih1[j]        + bhh1[j];
  const float c1z  = bih1[64 + j]   + bhh1[64 + j];
  const float c1xn = bih1[128 + j];
  const float c1hn = bhh1[128 + j];
  const float fw = fcw[j];
  const float fb = fcb[0];

  // ---- WMMA B-fragments for all weights, pinned in VGPRs ----
  // B is 32x16 f16 per K-step: lanes 0-15 hold K=0..15 (2/VGPR), lanes 16-31 K=16..31.
  // B[k][n] = W[n][k] with n = gate*64 + j (gate-major row of the [192,64] weight).
  v16h wh0[3][2], wi1[3][2], wh1[3][2];
  {
    const int koffl = hf * 16;
#pragma unroll
    for (int g = 0; g < 3; ++g) {
      const int n = g * 64 + j;
#pragma unroll
      for (int s = 0; s < 2; ++s) {
        const int kb = 32 * s + koffl;
#pragma unroll
        for (int v = 0; v < 8; ++v) {
          wh0[g][s][2 * v]     = (_Float16)Whh0[n * 64 + kb + 2 * v];
          wh0[g][s][2 * v + 1] = (_Float16)Whh0[n * 64 + kb + 2 * v + 1];
          wi1[g][s][2 * v]     = (_Float16)Wih1[n * 64 + kb + 2 * v];
          wi1[g][s][2 * v + 1] = (_Float16)Wih1[n * 64 + kb + 2 * v + 1];
          wh1[g][s][2 * v]     = (_Float16)Whh1[n * 64 + kb + 2 * v];
          wh1[g][s][2 * v + 1] = (_Float16)Whh1[n * 64 + kb + 2 * v + 1];
        }
      }
    }
  }

  // x prefetch register (wave 0): lane -> (row = lane/2, component = lane&1)
  float xreg = 0.f;
  if (wav == 0) {
    const int row = lane >> 1, c = lane & 1;
    xbuf[0][lane] = x[((size_t)(m0 + row) * T_SEQ + 0) * 2 + c];  // x_0
    xreg = x[((size_t)(m0 + row) * T_SEQ + 1) * 2 + c];           // x_1
  }

  // persistent f32 hidden state, elementwise-aligned with WMMA C/D layout:
  // component i of lane -> (batch row i + 8*hf, hidden col j)
  v8f h1l = {};
  v8f h2l = {};

  __syncthreads();

  // A-fragment loader (16x32 f16): lane<16: K=0..15; lane>=16: K=16..31
  auto ldA = [&](const _Float16* hb, int s) -> v16h {
    const int base = (lane & 15) * LDH + 32 * s + hf * 16;
    v8h lo = *(const v8h*)(hb + base);
    v8h hi = *(const v8h*)(hb + base + 8);
    v16h a;
#pragma unroll
    for (int i = 0; i < 8; ++i) { a[i] = lo[i]; a[8 + i] = hi[i]; }
    return a;
  };

#pragma unroll 1
  for (int t = 0; t < T_SEQ; ++t) {
    const int cur = t & 1, prv = cur ^ 1;

    // ================= phase P =================
    // h1[t-1] fragments from buffer prv (written in P(t-1); sync(t-1) orders)
    v16h a1k0 = ldA(lds_h1[prv], 0), a1k1 = ldA(lds_h1[prv], 1);

    if (wav == 0) {
      xbuf[prv][lane] = xreg;     // publish x_{t+1} into buffer (t+1)&1
      const int row = lane >> 1, c = lane & 1;
      const int tn = (t + 2 < T_SEQ) ? t + 2 : t;
      xreg = x[((size_t)(m0 + row) * T_SEQ + tn) * 2 + c];  // prefetch x_{t+2}
      const int tp = (t + 8 < T_SEQ) ? t + 8 : t;
      __builtin_prefetch(&x[((size_t)(m0 + row) * T_SEQ + tp) * 2 + c], 0, 0);
    }

    // ---------------- layer 0 ----------------
    float x0v[8], x1v[8];
#pragma unroll
    for (int i = 0; i < 8; ++i) {
      const int r = i + 8 * hf;
      x0v[i] = xbuf[cur][2 * r];        // x_t (published in P(t-1))
      x1v[i] = xbuf[cur][2 * r + 1];
    }
    v8f aR, aZ, aHN;
    float xn0[8];
#pragma unroll
    for (int i = 0; i < 8; ++i) {
      aR[i]  = fmaf(x0v[i], wr0, fmaf(x1v[i], wr1, c0r));
      aZ[i]  = fmaf(x0v[i], wz0, fmaf(x1v[i], wz1, c0z));
      xn0[i] = fmaf(x0v[i], wn0, fmaf(x1v[i], wn1, c0xn));
      aHN[i] = c0hn;
    }
    aR  = wmma16(a1k0, wh0[0][0], aR);  aR  = wmma16(a1k1, wh0[0][1], aR);
    aZ  = wmma16(a1k0, wh0[1][0], aZ);  aZ  = wmma16(a1k1, wh0[1][1], aZ);
    aHN = wmma16(a1k0, wh0[2][0], aHN); aHN = wmma16(a1k1, wh0[2][1], aHN);
#pragma unroll
    for (int i = 0; i < 8; ++i) {
      const float r = sigf(aR[i]);
      const float z = sigf(aZ[i]);
      const float n = dtanh(fmaf(r, aHN[i], xn0[i]));
      h1l[i] = (1.f - z) * n + z * h1l[i];
    }
#pragma unroll
    for (int i = 0; i < 8; ++i)
      lds_h1[cur][(i + 8 * hf) * LDH + j] = (_Float16)h1l[i];

    __syncthreads();   // single barrier per timestep: h1[t] published

    // ================= phase Q =================
    // drain previous step's head result (adds in Q(t-1), ordered by sync(t))
    if (t > 0 && tid < 16) {
      out[(size_t)(m0 + tid) * T_SEQ + (t - 1)] = outacc[prv][tid] + fb;
      outacc[prv][tid] = 0.f;   // next adds to prv happen in Q(t+1), after sync(t+1)
    }

    // ---------------- layer 1 ----------------
    v16h a2k0 = ldA(lds_h1[cur], 0), a2k1 = ldA(lds_h1[cur], 1);  // h1[t]
    v16h a3k0 = ldA(lds_h2[prv], 0), a3k1 = ldA(lds_h2[prv], 1);  // h2[t-1]
    v8f bR, bZ, bXN, bHN;
#pragma unroll
    for (int i = 0; i < 8; ++i) {
      bR[i] = c1r; bZ[i] = c1z; bXN[i] = c1xn; bHN[i] = c1hn;
    }
    bR  = wmma16(a3k0, wh1[0][0], bR);  bR  = wmma16(a3k1, wh1[0][1], bR);
    bZ  = wmma16(a3k0, wh1[1][0], bZ);  bZ  = wmma16(a3k1, wh1[1][1], bZ);
    bHN = wmma16(a3k0, wh1[2][0], bHN); bHN = wmma16(a3k1, wh1[2][1], bHN);
    bR  = wmma16(a2k0, wi1[0][0], bR);  bR  = wmma16(a2k1, wi1[0][1], bR);
    bZ  = wmma16(a2k0, wi1[1][0], bZ);  bZ  = wmma16(a2k1, wi1[1][1], bZ);
    bXN = wmma16(a2k0, wi1[2][0], bXN); bXN = wmma16(a2k1, wi1[2][1], bXN);

    float p[8];
#pragma unroll
    for (int i = 0; i < 8; ++i) {
      const float r = sigf(bR[i]);
      const float z = sigf(bZ[i]);
      const float n = dtanh(fmaf(r, bHN[i], bXN[i]));
      h2l[i] = (1.f - z) * n + z * h2l[i];
      p[i] = h2l[i] * fw;                       // head partial for col j
    }
#pragma unroll
    for (int i = 0; i < 8; ++i)
      lds_h2[cur][(i + 8 * hf) * LDH + j] = (_Float16)h2l[i];

    // head: butterfly-reduce over the 16 column-lanes of each half
#pragma unroll
    for (int m = 1; m < 16; m <<= 1) {
#pragma unroll
      for (int i = 0; i < 8; ++i) p[i] += __shfl_xor(p[i], m, 16);
    }
    if (lcol == 0) {
#pragma unroll
      for (int i = 0; i < 8; ++i) atomicAdd(&outacc[cur][i + 8 * hf], p[i]);
    }
    // no end-of-loop barrier: next iteration's sync orders all consumers
  }

  __syncthreads();
  if (tid < 16)
    out[(size_t)(m0 + tid) * T_SEQ + (T_SEQ - 1)] =
        outacc[(T_SEQ - 1) & 1][tid] + fb;
}

extern "C" void kernel_launch(void* const* d_in, const int* in_sizes, int n_in,
                              void* d_out, int out_size, void* d_ws, size_t ws_size,
                              hipStream_t stream) {
  (void)in_sizes; (void)n_in; (void)out_size; (void)d_ws; (void)ws_size;
  const float* x    = (const float*)d_in[0];
  const float* Wih0 = (const float*)d_in[1];
  const float* Whh0 = (const float*)d_in[2];
  const float* bih0 = (const float*)d_in[3];
  const float* bhh0 = (const float*)d_in[4];
  const float* Wih1 = (const float*)d_in[5];
  const float* Whh1 = (const float*)d_in[6];
  const float* bih1 = (const float*)d_in[7];
  const float* bhh1 = (const float*)d_in[8];
  const float* fcw  = (const float*)d_in[9];
  const float* fcb  = (const float*)d_in[10];
  float* out = (float*)d_out;

  dim3 grid(8);    // B/16 = 128/16 workgroups, one 16-row WMMA tile each
  dim3 block(128); // 4 waves (wave32): wave w owns gate columns 16w..16w+15
  gru2_fused_wmma<<<grid, block, 0, stream>>>(x, Wih0, Whh0, bih0, bhh0,
                                              Wih1, Whh1, bih1, bhh1,
                                              fcw, fcb, out);
}
